// Attention_62663572848863
// MI455X (gfx1250) — matrix operand order repaired and checked by reading, not compiled, not run on hardware
//
#include <hip/hip_runtime.h>
#include <hip/hip_bf16.h>

typedef _Float16 half_t;
typedef __attribute__((ext_vector_type(16))) _Float16 v16h;
typedef __attribute__((ext_vector_type(8)))  _Float16 v8h;
typedef __attribute__((ext_vector_type(4)))  _Float16 v4h;
typedef __attribute__((ext_vector_type(8)))  float    v8f;
typedef __attribute__((ext_vector_type(4)))  float    v4f;

constexpr int Bn = 4, Tn = 2048, Cn = 1024, Hn = 16, Dn = 64;
constexpr int NO = 3 * Cn;        // 3072 output channels of QKV proj
constexpr int Mn = Bn * Tn;       // 8192 rows

// ---------------------------------------------------------------------------
// f32 -> f16 conversion, 4 elements/thread (b128 load, b64 store)
// ---------------------------------------------------------------------------
__global__ __launch_bounds__(256) void cvt4(const float* __restrict__ s,
                                            half_t* __restrict__ d, int n4) {
  int i = blockIdx.x * blockDim.x + threadIdx.x;
  if (i < n4) {
    v4f f = ((const v4f*)s)[i];
    v4h h = { (half_t)f[0], (half_t)f[1], (half_t)f[2], (half_t)f[3] };
    ((v4h*)d)[i] = h;
  }
}

// ---------------------------------------------------------------------------
// QKV GEMM: qkv[m, o] = sum_c x[m,c] * W[o,c] + b[o]
//   M=8192, N=3072, K=1024.  One wave computes a 16x64 strip (4 WMMA tiles).
// Writes:  Q [B,H,T,D] f16 (pre-scaled by 1/sqrt(D))
//          K [B,H,T,D] f16
//          V [B,H,D,T] f16 (transposed so P*V B-operand loads are contiguous)
// ---------------------------------------------------------------------------
__global__ __launch_bounds__(256)
void qkv_gemm(const half_t* __restrict__ Xh, const half_t* __restrict__ Wh,
              const float* __restrict__ bias,
              half_t* __restrict__ Qh, half_t* __restrict__ Kh,
              half_t* __restrict__ Vt) {
  const int lane = threadIdx.x & 31;
  const int wave = threadIdx.x >> 5;
  const int hi = lane >> 4;        // which 16-lane half
  const int ln = lane & 15;

  const int m0 = blockIdx.x * 16;                 // row tile
  const int n0 = blockIdx.y * 512 + wave * 64;    // this wave's 64 channels

  // A operand (x rows), per-lane row = m0 + ln; k-chunks {0..7},{8..15} (+16 for hi)
  const half_t* arow = Xh + (size_t)(m0 + ln) * Cn;
  const int aoff = hi * 16;

  v8f acc[4] = {v8f{}, v8f{}, v8f{}, v8f{}};

  for (int kb = 0; kb < Cn; kb += 32) {
    v16h a;
    {
      v8h a0 = *(const v8h*)(arow + kb + aoff);
      v8h a1 = *(const v8h*)(arow + kb + aoff + 8);
#pragma unroll
      for (int i = 0; i < 8; ++i) { a[i] = a0[i]; a[i + 8] = a1[i]; }
    }
#pragma unroll
    for (int t = 0; t < 4; ++t) {
      // B operand: B[kk,n] = W[n0+t*16+n, kb+kk]; lane holds col n=ln,
      // halves = kk 0..15 (hi=0) or 16..31 (hi=1) -> contiguous 32B
      const half_t* brow = Wh + (size_t)(n0 + t * 16 + ln) * Cn + kb + hi * 16;
      v16h bm = *(const v16h*)brow;
      acc[t] = __builtin_amdgcn_wmma_f32_16x16x32_f16(
          false, a, false, bm, (short)0, acc[t], false, false);
    }
  }

  // Epilogue: bias + scatter.  C layout: vgpr r -> row m0 + r + 8*hi, col = ln.
#pragma unroll
  for (int t = 0; t < 4; ++t) {
    const int o = n0 + t * 16 + ln;   // output channel (uniform q/k/v per tile)
    const float bv = bias[o];
    const int which = o >> 10;        // 0=q, 1=k, 2=v
    const int rem = o & 1023;
    const int h = rem >> 6;
    const int d = rem & 63;
#pragma unroll
    for (int r = 0; r < 8; ++r) {
      const int m = m0 + r + hi * 8;
      const int bb = m >> 11;         // / 2048
      const int tt = m & 2047;
      const float val = acc[t][r] + bv;
      if (which == 0) {
        Qh[(((size_t)bb * Hn + h) * Tn + tt) * Dn + d] = (half_t)(val * 0.125f);
      } else if (which == 1) {
        Kh[(((size_t)bb * Hn + h) * Tn + tt) * Dn + d] = (half_t)val;
      } else {
        Vt[(((size_t)bb * Hn + h) * Dn + d) * Tn + tt] = (half_t)val;
      }
    }
  }
}

// ---------------------------------------------------------------------------
// Flash attention: one wave owns a 16-row query tile; online softmax; causal.
// ---------------------------------------------------------------------------
constexpr int FA_WAVES = 4;

__global__ __launch_bounds__(FA_WAVES * 32)
void flash_attn(const half_t* __restrict__ Qh, const half_t* __restrict__ Kh,
                const half_t* __restrict__ Vt, float* __restrict__ Out) {
  __shared__ __align__(16) half_t pshare[FA_WAVES][16][32];

  const int lane = threadIdx.x & 31;
  const int wave = threadIdx.x >> 5;
  const int hi = lane >> 4;
  const int ln = lane & 15;

  const int tile = blockIdx.x * FA_WAVES + wave;  // 0 .. B*H*(T/16)-1
  const int qt = tile & (Tn / 16 - 1);
  const int bh = tile / (Tn / 16);
  const int h  = bh & (Hn - 1);
  const int bb = bh / Hn;
  const int tq0 = qt * 16;

  const half_t* qb  = Qh + ((size_t)bb * Hn + h) * Tn * Dn;
  const half_t* kb_ = Kh + ((size_t)bb * Hn + h) * Tn * Dn;
  const half_t* vtb = Vt + ((size_t)bb * Hn + h) * Dn * Tn;

  // Q as two A-tiles (k-chunks over D=64), resident for the whole loop
  v16h aq[2];
  {
    const half_t* qrow = qb + (size_t)(tq0 + ln) * Dn;
#pragma unroll
    for (int c = 0; c < 2; ++c) {
      const int base = c * 32 + hi * 16;
      v8h a0 = *(const v8h*)(qrow + base);
      v8h a1 = *(const v8h*)(qrow + base + 8);
#pragma unroll
      for (int i = 0; i < 8; ++i) { aq[c][i] = a0[i]; aq[c][i + 8] = a1[i]; }
    }
  }

  v8f o_acc[4] = {v8f{}, v8f{}, v8f{}, v8f{}};  // 4 d-tiles of 16 cols
  float mst[8], lst[8];                          // per-row stats (row = r+8*hi)
#pragma unroll
  for (int r = 0; r < 8; ++r) { mst[r] = -1e30f; lst[r] = 0.0f; }

  const int nkb = (tq0 + 16 + 31) / 32;          // causal: only needed blocks
  for (int j = 0; j < nkb; ++j) {
    const int s0 = j * 32;

    // ---- S = Q K^T (two 16x16 col tiles, 2 k-steps each) ----
    v8f sc[2] = {v8f{}, v8f{}};
#pragma unroll
    for (int tc = 0; tc < 2; ++tc) {
      const half_t* krow = kb_ + (size_t)(s0 + tc * 16 + ln) * Dn;
#pragma unroll
      for (int c = 0; c < 2; ++c) {
        v16h bk = *(const v16h*)(krow + c * 32 + hi * 16);
        sc[tc] = __builtin_amdgcn_wmma_f32_16x16x32_f16(
            false, aq[c], false, bk, (short)0, sc[tc], false, false);
      }
    }

    // ---- causal mask + online softmax (rows live in one 16-lane half) ----
#pragma unroll
    for (int r = 0; r < 8; ++r) {
      const int row = tq0 + r + hi * 8;
      float v0 = (s0 + ln)      <= row ? sc[0][r] : -1e30f;
      float v1 = (s0 + 16 + ln) <= row ? sc[1][r] : -1e30f;

      float mx = fmaxf(v0, v1);
#pragma unroll
      for (int off = 8; off >= 1; off >>= 1)
        mx = fmaxf(mx, __shfl_xor(mx, off, 16));
      const float mnew  = fmaxf(mst[r], mx);
      const float alpha = __expf(mst[r] - mnew);
      mst[r] = mnew;

      const float p0 = __expf(v0 - mnew);
      const float p1 = __expf(v1 - mnew);
      float ps = p0 + p1;
#pragma unroll
      for (int off = 8; off >= 1; off >>= 1)
        ps += __shfl_xor(ps, off, 16);
      lst[r] = lst[r] * alpha + ps;

      o_acc[0][r] *= alpha; o_acc[1][r] *= alpha;
      o_acc[2][r] *= alpha; o_acc[3][r] *= alpha;

      pshare[wave][r + hi * 8][ln]      = (half_t)p0;
      pshare[wave][r + hi * 8][16 + ln] = (half_t)p1;
    }

    // Same-wave DS ops are hardware-in-order; just stop compiler reordering.
    asm volatile("" ::: "memory");

    // ---- reload P in A layout (16x32 over the 32 keys) ----
    v16h ap;
    {
      const half_t* prow = &pshare[wave][ln][0];
      const int base = hi * 16;
      v8h p0 = *(const v8h*)(prow + base);
      v8h p1 = *(const v8h*)(prow + base + 8);
#pragma unroll
      for (int i = 0; i < 8; ++i) { ap[i] = p0[i]; ap[i + 8] = p1[i]; }
    }
    asm volatile("" ::: "memory");

    // ---- O += P * V  (B operand contiguous thanks to V^T layout) ----
#pragma unroll
    for (int dt = 0; dt < 4; ++dt) {
      const half_t* vrow = vtb + (size_t)(dt * 16 + ln) * Tn + s0 + hi * 16;
      v16h bv = *(const v16h*)vrow;
      o_acc[dt] = __builtin_amdgcn_wmma_f32_16x16x32_f16(
          false, ap, false, bv, (short)0, o_acc[dt], false, false);
    }
  }

  // ---- finalize: O /= l, write [B,T,H,D] fp32 ----
#pragma unroll
  for (int r = 0; r < 8; ++r) {
    const float inv = 1.0f / lst[r];
    const int row = tq0 + r + hi * 8;
    float* orow = Out + ((size_t)bb * Tn + row) * Cn + h * Dn;
#pragma unroll
    for (int dt = 0; dt < 4; ++dt)
      orow[dt * 16 + ln] = o_acc[dt][r] * inv;
  }
}

// ---------------------------------------------------------------------------
extern "C" void kernel_launch(void* const* d_in, const int* in_sizes, int n_in,
                              void* d_out, int out_size, void* d_ws, size_t ws_size,
                              hipStream_t stream) {
  const float* x    = (const float*)d_in[0];
  const float* W    = (const float*)d_in[1];
  const float* bias = (const float*)d_in[2];
  float* out = (float*)d_out;

  char* ws = (char*)d_ws;
  size_t off = 0;
  half_t* Xh = (half_t*)(ws + off); off += (size_t)Mn * Cn * sizeof(half_t);
  half_t* Wh = (half_t*)(ws + off); off += (size_t)NO * Cn * sizeof(half_t);
  half_t* Qh = (half_t*)(ws + off); off += (size_t)Bn * Hn * Tn * Dn * sizeof(half_t);
  half_t* Kh = (half_t*)(ws + off); off += (size_t)Bn * Hn * Tn * Dn * sizeof(half_t);
  half_t* Vt = (half_t*)(ws + off); off += (size_t)Bn * Hn * Tn * Dn * sizeof(half_t);

  {
    const int n4 = Mn * Cn / 4;
    cvt4<<<n4 / 256, 256, 0, stream>>>(x, Xh, n4);
  }
  {
    const int n4 = NO * Cn / 4;
    cvt4<<<n4 / 256, 256, 0, stream>>>(W, Wh, n4);
  }

  dim3 g2(Mn / 16, NO / 512);
  qkv_gemm<<<g2, 256, 0, stream>>>(Xh, Wh, bias, Qh, Kh, Vt);

  const int tiles = Bn * Hn * (Tn / 16);       // 8192
  flash_attn<<<tiles / FA_WAVES, FA_WAVES * 32, 0, stream>>>(Qh, Kh, Vt, out);
}
